// Model_12506944766462
// MI455X (gfx1250) — matrix operand order repaired and checked by reading, not compiled, not run on hardware
//
#include <hip/hip_runtime.h>

typedef __attribute__((ext_vector_type(16))) _Float16 v16h;
typedef __attribute__((ext_vector_type(8)))  _Float16 v8h;
typedef __attribute__((ext_vector_type(2)))  _Float16 v2h;
typedef __attribute__((ext_vector_type(8)))  float    v8f;

#define B_   16
#define T_   384
#define P_   100
#define A_   3
#define D_   768
#define H_   32
#define W_   9
#define BT_  (B_ * T_)
#define DIN_ 600
#define D2_  1536

#define BM 128
#define BN 128
#define BK 32

// ---------------------------------------------------------------------------
// Convert activation A[M,K] f32 -> A16[M,Kp] f16 (zero-padded K), paired
// ---------------------------------------------------------------------------
__global__ __launch_bounds__(256)
void aconv(const float* __restrict__ A, _Float16* __restrict__ A16,
           int M, int K, int Kp)
{
    size_t idx = (size_t)blockIdx.x * 256 + threadIdx.x;
    size_t total = (size_t)M * (Kp >> 1);
    if (idx >= total) return;
    int m = (int)(idx / (Kp >> 1));
    int k = (int)(idx % (Kp >> 1)) * 2;
    const float* row = A + (size_t)m * K;
    v2h o;
    o[0] = (k     < K) ? (_Float16)row[k]     : (_Float16)0.f;
    o[1] = (k + 1 < K) ? (_Float16)row[k + 1] : (_Float16)0.f;
    ((v2h*)A16)[idx] = o;
}

// ---------------------------------------------------------------------------
// Convert+transpose weight W[K,N] f32 -> Wt[Np,Kp] f16 (zero-padded)
// ---------------------------------------------------------------------------
__global__ __launch_bounds__(256)
void wconv(const float* __restrict__ W, _Float16* __restrict__ Wt,
           int K, int N, int Kp, int Np)
{
    size_t idx = (size_t)blockIdx.x * 256 + threadIdx.x;
    size_t total = (size_t)Np * Kp;
    if (idx >= total) return;
    int k = (int)(idx / Np), n = (int)(idx % Np);   // coalesced read over n
    float v = (k < K && n < N) ? W[(size_t)k * N + n] : 0.f;
    Wt[(size_t)n * Kp + k] = (_Float16)v;
}

// ---------------------------------------------------------------------------
// WMMA GEMM, LDS-free, Kp compile-time: C[M,N] = A16[M,KP] @ Wt[Np,KP]^T + bias
// Block = 256 threads (8 waves); wave w owns rows [w*16,w*16+16) x 128 cols.
// 2-stage software pipeline over K; all tile offsets fold into IOFFSET.
// ---------------------------------------------------------------------------
template <int KP>
__global__ __launch_bounds__(256)
void gemm_wmma(const _Float16* __restrict__ A16, const _Float16* __restrict__ Wt,
               const float* __restrict__ bias, float* __restrict__ Cmat,
               int M, int N)
{
    const int tid  = threadIdx.x;
    const int wave = tid >> 5;
    const int lane = tid & 31;
    const int m0   = blockIdx.y * BM;
    const int n0   = blockIdx.x * BN;

    v8f acc[8];
#pragma unroll
    for (int c = 0; c < 8; ++c) acc[c] = (v8f){0.f,0.f,0.f,0.f,0.f,0.f,0.f,0.f};

    // A fragment: lane l -> row = wave*16 + (l&15); khalf = (l>>4)*8
    const int mrow  = m0 + wave * 16 + (lane & 15);
    const int mload = (mrow < M) ? mrow : 0;        // stores guarded below
    const _Float16* pa  = A16 + (size_t)mload * KP + ((lane >> 4) * 8);
    // B fragments: lane l -> col = c*16 + (l&15); khalf = (l>>4)*8
    const _Float16* pb0 = Wt + (size_t)(n0 + (lane & 15)) * KP + ((lane >> 4) * 8);

    auto loadA = [&](int k0) -> v16h {
        v8h alo = *(const v8h*)(pa + k0);
        v8h ahi = *(const v8h*)(pa + k0 + 16);
        return __builtin_shufflevector(alo, ahi,
            0,1,2,3,4,5,6,7,8,9,10,11,12,13,14,15);
    };
    auto loadB = [&](int k0, int c) -> v16h {
        const _Float16* pb = pb0 + (size_t)c * 16 * KP + k0;   // const offset
        v8h blo = *(const v8h*)pb;
        v8h bhi = *(const v8h*)(pb + 16);
        return __builtin_shufflevector(blo, bhi,
            0,1,2,3,4,5,6,7,8,9,10,11,12,13,14,15);
    };

    // prologue: fragments for k0 = 0
    v16h afC = loadA(0);
    v16h bfC[8];
#pragma unroll
    for (int c = 0; c < 8; ++c) bfC[c] = loadB(0, c);

#pragma unroll
    for (int k0 = 0; k0 < KP; k0 += BK) {
        v16h afN;
        v16h bfN[8];
        if (k0 + BK < KP) {                 // compile-time per unrolled iter
            afN = loadA(k0 + BK);
#pragma unroll
            for (int c = 0; c < 8; ++c) bfN[c] = loadB(k0 + BK, c);
        }
#pragma unroll
        for (int c = 0; c < 8; ++c) {
            acc[c] = __builtin_amdgcn_wmma_f32_16x16x32_f16(
                false, afC, false, bfC[c], (short)0, acc[c], false, false);
        }
        if (k0 + BK < KP) {
            afC = afN;
#pragma unroll
            for (int c = 0; c < 8; ++c) bfC[c] = bfN[c];
        }
    }

    // ---- epilogue ----
    // lane l -> col = c*16 + (l&15); VGPR r -> row = wave*16 + (l>>4)*8 + r
    const int rbase = m0 + wave * 16 + ((lane >> 4) * 8);
    const int cbase = n0 + (lane & 15);

    if (rbase + 8 <= M && n0 + BN <= N) {
        // fast path: full tile, no guards, incremental addressing
        float bv[8];
#pragma unroll
        for (int c = 0; c < 8; ++c) bv[c] = bias ? bias[cbase + c * 16] : 0.f;
        float* pc = Cmat + (size_t)rbase * N + cbase;
#pragma unroll
        for (int r = 0; r < 8; ++r) {
#pragma unroll
            for (int c = 0; c < 8; ++c) pc[c * 16] = acc[c][r] + bv[c];
            pc += N;
        }
    } else {
#pragma unroll
        for (int c = 0; c < 8; ++c) {
#pragma unroll
            for (int r = 0; r < 8; ++r) {
                int row = rbase + r;
                int col = cbase + c * 16;
                if (row < M && col < N) {
                    float v = acc[c][r];
                    if (bias) v += bias[col];
                    Cmat[(size_t)row * N + col] = v;
                }
            }
        }
    }
}

// ---------------------------------------------------------------------------
// Global axis-LN stats over (T,P) per (b,a):  mean and 1/(std+1e-5)
// ---------------------------------------------------------------------------
__global__ __launch_bounds__(256)
void kstats(const float* __restrict__ x, float* __restrict__ meanBA,
            float* __restrict__ scaleBA)
{
    int b = blockIdx.x / A_;
    int a = blockIdx.x % A_;
    float sw = 0.f, s1 = 0.f, s2 = 0.f;
    for (int i = threadIdx.x; i < T_ * P_; i += 256) {
        int t = i / P_, p = i % P_;
        float v = x[(((size_t)b * T_ + t) * P_ + p) * A_ + a];
        if (v != 0.f) { sw += 1.f; s1 += v; s2 += v * v; }
    }
    __shared__ float r0[256], r1[256], r2[256];
    r0[threadIdx.x] = sw; r1[threadIdx.x] = s1; r2[threadIdx.x] = s2;
    __syncthreads();
    for (int s = 128; s > 0; s >>= 1) {
        if (threadIdx.x < s) {
            r0[threadIdx.x] += r0[threadIdx.x + s];
            r1[threadIdx.x] += r1[threadIdx.x + s];
            r2[threadIdx.x] += r2[threadIdx.x + s];
        }
        __syncthreads();
    }
    if (threadIdx.x == 0) {
        float denom = r0[0] + 1e-5f;
        float m = r1[0] / denom;
        float var = r2[0] / denom - 2.f * m * r1[0] / denom + m * m * r0[0] / denom;
        float sd = sqrtf(fmaxf(var, 0.f));
        meanBA[blockIdx.x]  = m;
        scaleBA[blockIdx.x] = 1.f / (sd + 1e-5f);
    }
}

struct FnParams { const float* g[4]; const float* b[4]; };

// ---------------------------------------------------------------------------
// Assemble h0[bt, 600] = concat(nx flattened, per-range feats) ; notmask[bt]
// ---------------------------------------------------------------------------
__global__ __launch_bounds__(128)
void assemble(const float* __restrict__ x, const float* __restrict__ xg,
              const float* __restrict__ xb, FnParams fn,
              const float* __restrict__ meanBA, const float* __restrict__ scaleBA,
              float* __restrict__ h0, float* __restrict__ notmask)
{
    const int starts[4] = {0, 21, 42, 82};
    const int ends[4]   = {21, 42, 82, 100};
    int bt = blockIdx.x;
    int b  = bt / T_;
    __shared__ float fm[12], fsc[12], fwc[12];
    int tid = threadIdx.x;
    if (tid < 12) {
        int a = tid >> 2, r = tid & 3;
        float sw = 0.f, s1 = 0.f, s2 = 0.f;
        for (int p = starts[r]; p < ends[r]; ++p) {
            float v = x[((size_t)bt * P_ + p) * A_ + a];
            if (v != 0.f) { sw += 1.f; s1 += v; s2 += v * v; }
        }
        float denom = sw + 1e-5f;
        float m = s1 / denom;
        float var = s2 / denom - 2.f * m * s1 / denom + m * m * sw / denom;
        float sd = sqrtf(fmaxf(var, 0.f));
        fm[tid] = m; fsc[tid] = 1.f / (sd + 1e-5f); fwc[tid] = sw;
    }
    __syncthreads();
    if (tid == 0) {
        float c = 0.f;
        for (int k = 0; k < 12; ++k) c += fwc[k];
        notmask[bt] = (c > 0.f) ? 1.f : 0.f;
    }
    for (int i = tid; i < 600; i += 128) {
        float outv;
        if (i < 300) {
            int p = i / 3, a = i % 3;
            float v = x[((size_t)bt * P_ + p) * A_ + a];
            float w = (v != 0.f) ? 1.f : 0.f;
            float m = meanBA[b * 3 + a], sc = scaleBA[b * 3 + a];
            outv = (xg[p * 3 + a] * ((v - m) * sc) + xb[p * 3 + a]) * w;
        } else {
            int q = i - 300;
            int r = 0, base = 0;
            if (q >= 246)      { r = 3; base = 246; }
            else if (q >= 126) { r = 2; base = 126; }
            else if (q >= 63)  { r = 1; base = 63;  }
            int ql = q - base;
            int pl = ql / 3, a = ql % 3;
            int p  = starts[r] + pl;
            float v = x[((size_t)bt * P_ + p) * A_ + a];
            float w = (v != 0.f) ? 1.f : 0.f;
            int si = a * 4 + r;
            outv = (fn.g[r][pl * 3 + a] * ((v - fm[si]) * fsc[si]) + fn.b[r][pl * 3 + a]) * w;
        }
        h0[(size_t)bt * 600 + i] = outv;
    }
}

// ---------------------------------------------------------------------------
// Row LayerNorm (+ELU, +posenc, +residual)
// ---------------------------------------------------------------------------
__global__ __launch_bounds__(256)
void rowln(const float* __restrict__ in, const float* __restrict__ g,
           const float* __restrict__ bb, const float* __restrict__ res,
           float* __restrict__ out, int n, int do_elu, int do_pos)
{
    int row = blockIdx.x;
    const float* xi = in + (size_t)row * n;
    float s = 0.f, s2 = 0.f;
    for (int i = threadIdx.x; i < n; i += 256) {
        float v = xi[i]; s += v; s2 += v * v;
    }
    __shared__ float r0[256], r1[256];
    r0[threadIdx.x] = s; r1[threadIdx.x] = s2;
    __syncthreads();
    for (int k = 128; k > 0; k >>= 1) {
        if (threadIdx.x < k) {
            r0[threadIdx.x] += r0[threadIdx.x + k];
            r1[threadIdx.x] += r1[threadIdx.x + k];
        }
        __syncthreads();
    }
    float mean = r0[0] / n;
    float var  = r1[0] / n - mean * mean;
    float rstd = 1.f / sqrtf(fmaxf(var, 0.f) + 1e-5f);
    int t = row % T_;
    for (int i = threadIdx.x; i < n; i += 256) {
        float v = g[i] * ((xi[i] - mean) * rstd) + bb[i];
        if (do_elu) v = (v > 0.f) ? v : (expf(v) - 1.f);
        if (do_pos) {
            int j = i >> 1;
            float dv = expf(-(logf(10000.f) / 768.f) * (float)(2 * j));
            float ang = (float)t * dv;
            v += (i & 1) ? cosf(ang) : sinf(ang);
        }
        if (res) v += res[(size_t)row * n + i];
        out[(size_t)row * n + i] = v;
    }
}

// ---------------------------------------------------------------------------
// attn_exp = exp(logit)*notmask ; attn = attn_exp / (colsum_T + 1e-5)
// ---------------------------------------------------------------------------
__global__ __launch_bounds__(256)
void attn_softmax(const float* __restrict__ logits, const float* __restrict__ notmask,
                  float* __restrict__ aexp, float* __restrict__ anorm)
{
    int b = blockIdx.x >> 5;
    int h = blockIdx.x & 31;
    float s = 0.f;
    for (int t = threadIdx.x; t < T_; t += 256) {
        size_t idx = (size_t)(b * T_ + t) * H_ + h;
        float e = expf(logits[idx]) * notmask[b * T_ + t];
        aexp[idx] = e;
        s += e;
    }
    __shared__ float red[256];
    red[threadIdx.x] = s;
    __syncthreads();
    for (int k = 128; k > 0; k >>= 1) {
        if (threadIdx.x < k) red[threadIdx.x] += red[threadIdx.x + k];
        __syncthreads();
    }
    float inv = 1.f / (red[0] + 1e-5f);
    for (int t = threadIdx.x; t < T_; t += 256) {
        size_t idx = (size_t)(b * T_ + t) * H_ + h;
        anorm[idx] = aexp[idx] * inv;
    }
}

// g_pool[b, d] = sum_t anorm[b,t,h(d)] * v[b,t,d]
__global__ __launch_bounds__(256)
void gpool_k(const float* __restrict__ anorm, const float* __restrict__ v,
             float* __restrict__ gp)
{
    int idx = blockIdx.x * 256 + threadIdx.x;
    if (idx >= B_ * D_) return;
    int b = idx / D_, d = idx % D_;
    int h = d / (D_ / H_);
    float acc = 0.f;
    for (int t = 0; t < T_; ++t)
        acc += anorm[(size_t)(b * T_ + t) * H_ + h] * v[(size_t)(b * T_ + t) * D_ + d];
    gp[idx] = acc;
}

__global__ __launch_bounds__(256)
void expk(const float* __restrict__ in, float* __restrict__ out, int n)
{
    int i = blockIdx.x * 256 + threadIdx.x;
    if (i < n) out[i] = expf(in[i]);
}

// ---------------------------------------------------------------------------
// Sliding-window combine (faithful to reference slot pairing)
// ---------------------------------------------------------------------------
__global__ __launch_bounds__(256)
void window_combine(const float* __restrict__ aexp, const float* __restrict__ posE,
                    const float* __restrict__ v, const float* __restrict__ gp,
                    float* __restrict__ ow, int dil)
{
    int bt = blockIdx.x;
    int b = bt / T_, t = bt % T_;
    __shared__ float aw[(W_ + 1) * H_];
    for (int i = threadIdx.x; i < (W_ + 1) * H_; i += 256) {
        int j = i >> 5, h = i & 31;
        float val;
        if (j == 0) {
            val = posE[(b * (W_ + 1) + 0) * H_ + h];
        } else {
            int w = j - 1;
            int idx = dil * w + t - dil * (W_ / 2);
            float ae = (idx >= 0 && idx < T_) ? aexp[(size_t)(b * T_ + idx) * H_ + h] : 0.f;
            val = ae * posE[(b * (W_ + 1) + j) * H_ + h];
        }
        aw[j * H_ + h] = val;
    }
    __syncthreads();
    if (threadIdx.x < H_) {
        float s = 0.f;
        for (int j = 0; j < W_ + 1; ++j) s += aw[j * H_ + threadIdx.x];
        float inv = 1.f / (s + 1e-5f);
        for (int j = 0; j < W_ + 1; ++j) aw[j * H_ + threadIdx.x] *= inv;
    }
    __syncthreads();
    for (int d = threadIdx.x; d < D_; d += 256) {
        int h = d / (D_ / H_);
        float acc = aw[0 * H_ + h] * gp[(size_t)b * D_ + d];
        for (int j = 0; j < W_; ++j) {
            int idx = dil * j + t - dil * (W_ / 2);
            float vv = (idx >= 0 && idx < T_) ? v[(size_t)(b * T_ + idx) * D_ + d] : 0.f;
            acc += aw[j * H_ + h] * vv;
        }
        ow[(size_t)bt * D_ + d] = acc;
    }
}

// ---------------------------------------------------------------------------
// Host side
// ---------------------------------------------------------------------------
static inline int padup(int x, int m) { return (x + m - 1) / m * m; }

extern "C" void kernel_launch(void* const* d_in, const int* in_sizes, int n_in,
                              void* d_out, int out_size, void* d_ws, size_t ws_size,
                              hipStream_t stream)
{
    (void)in_sizes; (void)n_in; (void)out_size; (void)ws_size;
    int i = 0;
    const float* x    = (const float*)d_in[i++];
    const float* xn_g = (const float*)d_in[i++];
    const float* xn_b = (const float*)d_in[i++];
    FnParams fnp;
    for (int r = 0; r < 4; ++r) {
        fnp.g[r] = (const float*)d_in[i++];
        fnp.b[r] = (const float*)d_in[i++];
    }
    const float* in1_w = (const float*)d_in[i++];
    const float* in1_b = (const float*)d_in[i++];
    const float* ln1_g = (const float*)d_in[i++];
    const float* ln1_b = (const float*)d_in[i++];
    const float* in2_w = (const float*)d_in[i++];
    const float* in2_b = (const float*)d_in[i++];
    const float* ln2_g = (const float*)d_in[i++];
    const float* ln2_b = (const float*)d_in[i++];
    struct BlkP {
        const float *aw, *ab, *p1w, *p1b, *plg, *plb, *p2w, *p2b;
        const float *vw, *vb, *vlg, *vlb, *ow, *ob, *olg, *olb;
    } blk[4];
    for (int k = 0; k < 4; ++k) {
        blk[k].aw  = (const float*)d_in[i++]; blk[k].ab  = (const float*)d_in[i++];
        blk[k].p1w = (const float*)d_in[i++]; blk[k].p1b = (const float*)d_in[i++];
        blk[k].plg = (const float*)d_in[i++]; blk[k].plb = (const float*)d_in[i++];
        blk[k].p2w = (const float*)d_in[i++]; blk[k].p2b = (const float*)d_in[i++];
        blk[k].vw  = (const float*)d_in[i++]; blk[k].vb  = (const float*)d_in[i++];
        blk[k].vlg = (const float*)d_in[i++]; blk[k].vlb = (const float*)d_in[i++];
        blk[k].ow  = (const float*)d_in[i++]; blk[k].ob  = (const float*)d_in[i++];
        blk[k].olg = (const float*)d_in[i++]; blk[k].olb = (const float*)d_in[i++];
    }
    const float* fo_w = (const float*)d_in[i++];
    const float* fo_b = (const float*)d_in[i++];

    // ---------------- workspace carving ----------------
    float* fbase = (float*)d_ws;
    auto carve = [&](size_t n) { float* p = fbase; fbase += n; return p; };
    float* meanBA  = carve(48);
    float* scaleBA = carve(48);
    float* notmask = carve(BT_);
    float* h0      = carve((size_t)BT_ * DIN_);
    float* hbuf    = carve((size_t)BT_ * D_);
    float* vbuf    = carve((size_t)BT_ * D_);
    float* t1      = carve((size_t)BT_ * D2_);
    float* t2      = carve((size_t)BT_ * D2_);
    float* aexp    = carve((size_t)BT_ * H_);
    float* anorm   = carve((size_t)BT_ * H_);
    float* gp      = carve((size_t)B_ * D_);
    float* gtmp    = carve(B_ * H_);
    float* h1p     = carve(B_ * H_);
    float* posb    = carve(B_ * (W_ + 1) * H_);
    float* posE    = carve(B_ * (W_ + 1) * H_);

    _Float16* hbase = (_Float16*)fbase;
    auto hcarve = [&](size_t n) { _Float16* p = hbase; hbase += n; return p; };

    const int KpDIN = padup(DIN_, BK);          // 608
    const int KpD   = D_;                       // 768
    const int KpD2  = D2_;                      // 1536
    const int KpH   = padup(H_, BK);            // 32
    const int NpH   = padup(H_, BN);            // 128
    const int NpD   = D_;                       // 768
    const int NpD2  = D2_;                      // 1536
    const int NpPOS = padup((W_ + 1) * H_, BN); // 384
    const int Np60  = padup(60, BN);            // 128

    _Float16* A16 = hcarve((size_t)BT_ * D2_);  // activation f16 staging (max size)
    _Float16* wt_in1 = hcarve((size_t)NpD2 * KpDIN);
    _Float16* wt_in2 = hcarve((size_t)NpD * KpD2);
    _Float16* wt_fo  = hcarve((size_t)Np60 * KpD);
    _Float16 *wt_attn[4], *wt_p1[4], *wt_p2[4], *wt_v[4], *wt_o[4];
    for (int k = 0; k < 4; ++k) {
        wt_attn[k] = hcarve((size_t)NpH * KpD);
        wt_p1[k]   = hcarve((size_t)NpH * KpD);
        wt_p2[k]   = hcarve((size_t)NpPOS * KpH);
        wt_v[k]    = hcarve((size_t)NpD * KpD);
        wt_o[k]    = hcarve((size_t)NpD * KpD);
    }

    const dim3 blk256(256);
    auto cgrid = [](size_t n) { return dim3((unsigned)((n + 255) / 256)); };

    auto launch_wconv = [&](const float* Wp, _Float16* Wt, int K, int N, int Kp, int Np) {
        wconv<<<cgrid((size_t)Np * Kp), blk256, 0, stream>>>(Wp, Wt, K, N, Kp, Np);
    };
    auto launch_aconv = [&](const float* Afp, int M, int K, int Kp) {
        aconv<<<cgrid((size_t)M * (Kp >> 1)), blk256, 0, stream>>>(Afp, A16, M, K, Kp);
    };
    auto launch_gemm = [&](const _Float16* Ah, const _Float16* Wth, const float* bias,
                           float* C, int M, int N, int Kp) {
        dim3 g((N + BN - 1) / BN, (M + BM - 1) / BM);
        switch (Kp) {
        case 32:   gemm_wmma<32><<<g, dim3(256), 0, stream>>>(Ah, Wth, bias, C, M, N); break;
        case 608:  gemm_wmma<608><<<g, dim3(256), 0, stream>>>(Ah, Wth, bias, C, M, N); break;
        case 768:  gemm_wmma<768><<<g, dim3(256), 0, stream>>>(Ah, Wth, bias, C, M, N); break;
        case 1536: gemm_wmma<1536><<<g, dim3(256), 0, stream>>>(Ah, Wth, bias, C, M, N); break;
        default: break;
        }
    };

    // ---- one-time weight conversion (f32 -> padded transposed f16) ----
    launch_wconv(in1_w, wt_in1, DIN_, D2_, KpDIN, NpD2);
    launch_wconv(in2_w, wt_in2, D2_, D_, KpD2, NpD);
    launch_wconv(fo_w,  wt_fo,  D_, 60, KpD, Np60);
    for (int k = 0; k < 4; ++k) {
        launch_wconv(blk[k].aw,  wt_attn[k], D_, H_, KpD, NpH);
        launch_wconv(blk[k].p1w, wt_p1[k],   D_, H_, KpD, NpH);
        launch_wconv(blk[k].p2w, wt_p2[k],   H_, (W_ + 1) * H_, KpH, NpPOS);
        launch_wconv(blk[k].vw,  wt_v[k],    D_, D_, KpD, NpD);
        launch_wconv(blk[k].ow,  wt_o[k],    D_, D_, KpD, NpD);
    }

    // ---- phase 0: features ----
    kstats<<<dim3(B_ * A_), blk256, 0, stream>>>(x, meanBA, scaleBA);
    assemble<<<dim3(BT_), dim3(128), 0, stream>>>(x, xn_g, xn_b, fnp, meanBA, scaleBA,
                                                  h0, notmask);

    // ---- input MLP ----
    launch_aconv(h0, BT_, DIN_, KpDIN);
    launch_gemm(A16, wt_in1, in1_b, t1, BT_, D2_, KpDIN);
    rowln<<<dim3(BT_), blk256, 0, stream>>>(t1, ln1_g, ln1_b, nullptr, t2, D2_, 1, 0);
    launch_aconv(t2, BT_, D2_, KpD2);
    launch_gemm(A16, wt_in2, in2_b, t1, BT_, D_, KpD2);
    rowln<<<dim3(BT_), blk256, 0, stream>>>(t1, ln2_g, ln2_b, nullptr, hbuf, D_, 1, 1);

    const int dils[4] = {1, 3, 3, 3};
    for (int k = 0; k < 4; ++k) {
        const BlkP& p = blk[k];
        // h -> f16 once; shared by attn and v GEMMs
        launch_aconv(hbuf, BT_, D_, KpD);
        launch_gemm(A16, wt_attn[k], p.ab, t1, BT_, H_, KpD);
        attn_softmax<<<dim3(B_ * H_), blk256, 0, stream>>>(t1, notmask, aexp, anorm);
        launch_gemm(A16, wt_v[k], p.vb, t2, BT_, D_, KpD);
        rowln<<<dim3(BT_), blk256, 0, stream>>>(t2, p.vlg, p.vlb, nullptr, vbuf, D_, 1, 0);
        // pooled global vector + pos MLP
        gpool_k<<<cgrid(B_ * D_), blk256, 0, stream>>>(anorm, vbuf, gp);
        launch_aconv(gp, B_, D_, KpD);
        launch_gemm(A16, wt_p1[k], p.p1b, gtmp, B_, H_, KpD);
        rowln<<<dim3(B_), blk256, 0, stream>>>(gtmp, p.plg, p.plb, nullptr, h1p, H_, 1, 0);
        launch_aconv(h1p, B_, H_, KpH);
        launch_gemm(A16, wt_p2[k], p.p2b, posb, B_, (W_ + 1) * H_, KpH);
        expk<<<cgrid(B_ * (W_ + 1) * H_), blk256, 0, stream>>>(posb, posE,
                                                               B_ * (W_ + 1) * H_);
        // window combine -> out projection -> ln/elu + residual
        window_combine<<<dim3(BT_), blk256, 0, stream>>>(aexp, posE, vbuf, gp, t2, dils[k]);
        launch_aconv(t2, BT_, D_, KpD);
        launch_gemm(A16, wt_o[k], p.ob, t1, BT_, D_, KpD);
        rowln<<<dim3(BT_), blk256, 0, stream>>>(t1, p.olg, p.olb, hbuf, hbuf, D_, 1, 0);
    }

    // ---- final projection 768 -> 60 into d_out ----
    launch_aconv(hbuf, BT_, D_, KpD);
    launch_gemm(A16, wt_fo, fo_b, (float*)d_out, BT_, 60, KpD);
}
